// AttSubLayerv2_36644660970134
// MI455X (gfx1250) — matrix operand order repaired and checked by reading, not compiled, not run on hardware
//
#include <hip/hip_runtime.h>
#include <math.h>

#define DEV __device__ __forceinline__
#define AS3 __attribute__((address_space(3)))

#if __has_builtin(__builtin_amdgcn_global_load_async_to_lds_b128) && \
    __has_builtin(__builtin_amdgcn_s_wait_asynccnt)
#define HAVE_ASYNC_LDS 1
#else
#define HAVE_ASYNC_LDS 0
#endif

typedef __attribute__((ext_vector_type(16))) __bf16 v16bf;
typedef __attribute__((ext_vector_type(8)))  float  v8f;
typedef __attribute__((ext_vector_type(4)))  int    v4i;

constexpr int B_ = 2, S_ = 2048, D_ = 256, H_ = 8, DK_ = 32;
constexpr float SCALE_ = 0.17677669529663687f; // 1/sqrt(32)

// ---------------- workspace layout (bytes) ----------------
constexpr size_t WS_RELT = 0;                                   // bf16 [B][S][S]  (relation^T per batch)
constexpr size_t WS_WQT  = WS_RELT + (size_t)B_ * S_ * S_ * 2;  // bf16 [256][256] (Wq^T)
constexpr size_t WS_WKT  = WS_WQT + 256 * 256 * 2;
constexpr size_t WS_WVT  = WS_WKT + 256 * 256 * 2;
constexpr size_t WS_WFT  = WS_WVT + 256 * 256 * 2;
constexpr size_t WS_Q    = WS_WFT + 256 * 256 * 2;              // bf16 [B][H][S][DK]
constexpr size_t WS_K    = WS_Q  + (size_t)B_ * H_ * S_ * DK_ * 2;
constexpr size_t WS_VT   = WS_K  + (size_t)B_ * H_ * S_ * DK_ * 2; // bf16 [B][H][DK][S]
constexpr size_t WS_ATT  = WS_VT + (size_t)B_ * H_ * S_ * DK_ * 2; // f32 [B*S][256]

// ---------------- bf16 helpers ----------------
DEV __bf16 to_bf16(float f) {
  unsigned u = __builtin_bit_cast(unsigned, f);
  unsigned r = u + 0x7FFFu + ((u >> 16) & 1u);   // round-to-nearest-even
  unsigned short h = (unsigned short)(r >> 16);
  return __builtin_bit_cast(__bf16, h);
}
DEV float bf16_to_f32(__bf16 x) {
  unsigned short h = __builtin_bit_cast(unsigned short, x);
  return __builtin_bit_cast(float, (unsigned)h << 16);
}

// ---------------- WMMA fragment helpers (wave32) ----------------
// A fragment: 16x32 bf16, element (m,k) at base[m*stride + k].
// Lanes 0-15 hold M=lane, K {0..15}; lanes 16-31 hold K {16..31} (contiguous, matches load_fragB).
DEV v16bf load_fragA(const __bf16* base, int stride, int lane) {
  int m  = lane & 15;
  int kb = (lane & 16) ? 16 : 0;
  const __bf16* row = base + (size_t)m * stride + kb;
  v16bf f;
#pragma unroll
  for (int i = 0; i < 4; ++i) {
    f[2 * i]     = row[2 * i];
    f[2 * i + 1] = row[2 * i + 1];
    f[8 + 2 * i] = row[8 + 2 * i];
    f[9 + 2 * i] = row[8 + 2 * i + 1];
  }
  return f;
}
// B fragment: 32x16 bf16 stored N-major (Bt[n][k]); lane holds column n=lane&15,
// contiguous K half selected by lane>=16 (per SWMMAC B layout analog).
DEV v16bf load_fragB(const __bf16* base, int stride, int lane) {
  int n  = lane & 15;
  int kb = (lane & 16) ? 16 : 0;
  const __bf16* row = base + (size_t)n * stride + kb;
  v16bf f;
#pragma unroll
  for (int i = 0; i < 16; ++i) f[i] = row[i];
  return f;
}
// C/D: lane L, vgpr r -> row = r + (L>=16 ? 8 : 0), col = L&15
DEV void store_frag_bf16(__bf16* base, int stride, int lane, v8f acc) {
  int n = lane & 15, mb = (lane & 16) ? 8 : 0;
#pragma unroll
  for (int r = 0; r < 8; ++r) base[(size_t)(mb + r) * stride + n] = to_bf16(acc[r]);
}

DEV v8f wmma_bf16(v16bf a, v16bf b, v8f c) {
  return __builtin_amdgcn_wmma_f32_16x16x32_bf16(false, a, false, b, (short)0, c,
                                                 false, false);
}

// ---------------- kernel 0: transpose + f32->bf16 convert ----------------
// out[c*R + r] = bf16(in[r*C + c]); grid.z batches with stride R*C.
__global__ __launch_bounds__(256) void transpose_to_bf16(
    const float* __restrict__ in, __bf16* __restrict__ out, int R, int C) {
  __shared__ float tile[32][33];
  size_t off = (size_t)blockIdx.z * R * C;
  in += off;
  out += off;
  int c0 = blockIdx.x * 32, r0 = blockIdx.y * 32;
  int x = threadIdx.x;
  for (int j = threadIdx.y; j < 32; j += 8)
    tile[j][x] = in[(size_t)(r0 + j) * C + c0 + x];
  __syncthreads();
  for (int j = threadIdx.y; j < 32; j += 8)
    out[(size_t)(c0 + j) * R + r0 + x] = to_bf16(tile[x][j]);
}

// ---------------- kernel 1: QKV projection GEMM ----------------
// x [4096,256] f32 @ W^T(bf16,[n][k]) -> bf16 heads.
// vt_mode==0: dst[((b*H+h)*S + s)*DK + dk] ; vt_mode==1: dst[((b*H+h)*DK + dk)*S + s]
__global__ __launch_bounds__(256) void proj_kernel(
    const float* __restrict__ x, const __bf16* __restrict__ Wt,
    __bf16* __restrict__ dst, int vt_mode) {
  __shared__ __bf16 At[32 * 256];
  int R0 = blockIdx.x * 32;
  int t = threadIdx.x;
  for (int i = t; i < 32 * 256; i += 256) {
    int r = i >> 8, c = i & 255;
    At[i] = to_bf16(x[(size_t)(R0 + r) * 256 + c]);
  }
  __syncthreads();
  int w = t >> 5, lane = t & 31;
  int wm = w & 1, wn = w >> 1;
  int nB = blockIdx.y * 128 + wn * 32;
  v8f acc[2] = {{}, {}};
  for (int kk = 0; kk < 256; kk += 32) {
    v16bf a  = load_fragA(At + wm * 16 * 256 + kk, 256, lane);
    v16bf b0 = load_fragB(Wt + (size_t)nB * 256 + kk, 256, lane);
    v16bf b1 = load_fragB(Wt + (size_t)(nB + 16) * 256 + kk, 256, lane);
    acc[0] = wmma_bf16(a, b0, acc[0]);
    acc[1] = wmma_bf16(a, b1, acc[1]);
  }
  int nl = lane & 15, mb = (lane & 16) ? 8 : 0;
#pragma unroll
  for (int ti = 0; ti < 2; ++ti) {
    int gcol = nB + ti * 16 + nl;
    int h = gcol >> 5, dk = gcol & 31;
#pragma unroll
    for (int r = 0; r < 8; ++r) {
      int grow = R0 + wm * 16 + mb + r;
      int b = grow >> 11, s = grow & 2047;
      if (!vt_mode)
        dst[((size_t)(b * H_ + h) * S_ + s) * DK_ + dk] = to_bf16(acc[ti][r]);
      else
        dst[((size_t)(b * H_ + h) * DK_ + dk) * S_ + s] = to_bf16(acc[ti][r]);
    }
  }
}

// ---------------- kernel 2: fused attention core ----------------
// Per (b,h,32-row q-tile): qk strip -> @relation -> mask/scale -> softmax -> @V.
// Dynamic LDS: two bf16 [32][2048] strips (256KB) + Q tile + reduction scratch.
__global__ __launch_bounds__(256) void attn_kernel(
    const __bf16* __restrict__ Q, const __bf16* __restrict__ Kmat,
    const __bf16* __restrict__ Vt, const __bf16* __restrict__ relT,
    const unsigned char* __restrict__ mask, float* __restrict__ attn_out,
    float* __restrict__ att_ws) {
  extern __shared__ char smem[];
  __bf16* Astrip = (__bf16*)smem;        // [32][2048] qk (bf16)
  __bf16* Pstrip = Astrip + 32 * 2048;   // [32][2048] scores/attn (bf16)
  __bf16* qt     = Pstrip + 32 * 2048;   // [32][32]
  float* red     = (float*)(qt + 32 * 32);  // [32][8]
  float* rowmax  = red + 32 * 8;            // [32]
  float* rowsum  = rowmax + 32;             // [32]

  int q0 = blockIdx.x * 32;
  int h = blockIdx.y, b = blockIdx.z;
  int t = threadIdx.x, w = t >> 5, lane = t & 31;

  const __bf16* Qbh = Q + (size_t)(b * H_ + h) * S_ * DK_;
  const __bf16* Kbh = Kmat + (size_t)(b * H_ + h) * S_ * DK_;
  const __bf16* Vbh = Vt + (size_t)(b * H_ + h) * DK_ * S_;
  const __bf16* Rb = relT + (size_t)b * S_ * S_;
  const unsigned char* Mb = mask + (size_t)b * S_ * S_;

  // Q tile (32x32 bf16 = 2KB contiguous): async DMA global->LDS when available.
#if HAVE_ASYNC_LDS
  if (t < 128) {
    v4i* gsrc = (v4i*)(Qbh + (size_t)q0 * DK_ + (size_t)t * 8);
    v4i* ldst = (v4i*)(qt + (size_t)t * 8);
    __builtin_amdgcn_global_load_async_to_lds_b128(gsrc, ldst, 0, 0);
  }
  __builtin_amdgcn_s_wait_asynccnt(0);
#else
  for (int i = t; i < 32 * 32; i += 256) {
    int r = i >> 5, c = i & 31;
    qt[i] = Qbh[(size_t)(q0 + r) * DK_ + c];
  }
#endif
  __syncthreads();

  // phase 1: qk strip = Q_tile (32x32) @ K^T  (DK=32 = one WMMA K-step)
  v16bf a_lo = load_fragA(qt, 32, lane);
  v16bf a_hi = load_fragA(qt + 16 * 32, 32, lane);
  int nC = w * 256;  // this wave's 256-column chunk of the strip
  for (int i = 0; i < 16; ++i) {
    int n0 = nC + i * 16;
    v16bf bk = load_fragB(Kbh + (size_t)n0 * DK_, DK_, lane);  // K rows = N-major B
    v8f z = {};
    v8f c0 = wmma_bf16(a_lo, bk, z);
    v8f c1 = wmma_bf16(a_hi, bk, z);
    store_frag_bf16(Astrip + n0, 2048, lane, c0);
    store_frag_bf16(Astrip + 16 * 2048 + n0, 2048, lane, c1);
  }
  __syncthreads();

  // phase 2: scores = qk_strip @ relation (K-dim = 2048), 2 n-tiles per pass so
  // each pair of A-fragment LDS reads feeds 4 WMMAs.
  for (int i = 0; i < 8; ++i) {
    int n0 = nC + i * 32;
    if (i + 1 < 8) __builtin_prefetch(Rb + (size_t)(n0 + 32) * S_, 0, 0);
    v8f c00 = {}, c01 = {}, c10 = {}, c11 = {};
    for (int k0 = 0; k0 < S_; k0 += 32) {
      v16bf al = load_fragA(Astrip + k0, 2048, lane);
      v16bf ah = load_fragA(Astrip + 16 * 2048 + k0, 2048, lane);
      v16bf b0 = load_fragB(Rb + (size_t)n0 * S_ + k0, S_, lane);         // relT rows
      v16bf b1 = load_fragB(Rb + (size_t)(n0 + 16) * S_ + k0, S_, lane);
      c00 = wmma_bf16(al, b0, c00);
      c10 = wmma_bf16(ah, b0, c10);
      c01 = wmma_bf16(al, b1, c01);
      c11 = wmma_bf16(ah, b1, c11);
    }
    int nl = lane & 15, mb = (lane & 16) ? 8 : 0;
    v8f clo[2] = {c00, c01}, chi[2] = {c10, c11};
#pragma unroll
    for (int ti = 0; ti < 2; ++ti) {
      int nn = n0 + ti * 16 + nl;
#pragma unroll
      for (int r = 0; r < 8; ++r) {
        int m0 = mb + r;
        float s0 = clo[ti][r] * SCALE_;
        if (Mb[(size_t)(q0 + m0) * S_ + nn]) s0 = -1e9f;
        Pstrip[(size_t)m0 * 2048 + nn] = to_bf16(s0);
        int m1 = 16 + mb + r;
        float s1 = chi[ti][r] * SCALE_;
        if (Mb[(size_t)(q0 + m1) * S_ + nn]) s1 = -1e9f;
        Pstrip[(size_t)m1 * 2048 + nn] = to_bf16(s1);
      }
    }
  }
  __syncthreads();

  // softmax over each of 32 rows (2048 cols); 8 threads per row
  int row = t >> 3, seg = t & 7, cbase = seg * 256;
  float mx = -3.4e38f;
  for (int c = cbase; c < cbase + 256; ++c)
    mx = fmaxf(mx, bf16_to_f32(Pstrip[(size_t)row * 2048 + c]));
  red[row * 8 + seg] = mx;
  __syncthreads();
  if (t < 32) {
    float m = red[t * 8];
    for (int i = 1; i < 8; ++i) m = fmaxf(m, red[t * 8 + i]);
    rowmax[t] = m;
  }
  __syncthreads();
  float rm = rowmax[row], sum = 0.f;
  for (int c = cbase; c < cbase + 256; ++c) {
    float e = __expf(bf16_to_f32(Pstrip[(size_t)row * 2048 + c]) - rm);
    Pstrip[(size_t)row * 2048 + c] = to_bf16(e);
    sum += e;
  }
  red[row * 8 + seg] = sum;
  __syncthreads();
  if (t < 32) {
    float s = 0.f;
    for (int i = 0; i < 8; ++i) s += red[t * 8 + i];
    rowsum[t] = 1.0f / s;
  }
  __syncthreads();
  float inv = rowsum[row];
  float* ao = attn_out + ((size_t)(b * H_ + h) * S_ + (q0 + row)) * S_;
  for (int c = cbase; c < cbase + 256; ++c) {
    float p = bf16_to_f32(Pstrip[(size_t)row * 2048 + c]) * inv;
    Pstrip[(size_t)row * 2048 + c] = to_bf16(p);
    ao[c] = p;  // second reference output
  }
  __syncthreads();

  // phase 3: att = attn_strip (32x2048) @ V (2048x32); 4 waves, 1 tile each
  if (w < 4) {
    int msub = w & 1, nsub = w >> 1;
    v8f acc = {};
    for (int k0 = 0; k0 < S_; k0 += 32) {
      v16bf a = load_fragA(Pstrip + msub * 16 * 2048 + k0, 2048, lane);
      v16bf bv = load_fragB(Vbh + (size_t)(nsub * 16) * S_ + k0, S_, lane);
      acc = wmma_bf16(a, bv, acc);
    }
    int nl = lane & 15, mb = (lane & 16) ? 8 : 0;
#pragma unroll
    for (int r = 0; r < 8; ++r) {
      int grow = b * S_ + q0 + msub * 16 + mb + r;
      int gcol = h * 32 + nsub * 16 + nl;
      att_ws[(size_t)grow * 256 + gcol] = acc[r];
    }
  }
}

// ---------------- kernel 3: FC GEMM + bias + residual + LayerNorm ----------------
__global__ __launch_bounds__(256) void fc_ln_kernel(
    const float* __restrict__ att, const __bf16* __restrict__ Wt,
    const float* __restrict__ bfc, const float* __restrict__ query,
    const float* __restrict__ gamma, const float* __restrict__ beta,
    float* __restrict__ out) {
  __shared__ __bf16 At[32 * 256];
  __shared__ float res[32 * 256];
  __shared__ float red[32 * 8], red2[32 * 8];
  __shared__ float mu[32], rs[32];
  int R0 = blockIdx.x * 32;
  int t = threadIdx.x;
  for (int i = t; i < 32 * 256; i += 256) {
    int r = i >> 8, c = i & 255;
    At[i] = to_bf16(att[(size_t)(R0 + r) * 256 + c]);
  }
  __syncthreads();
  int w = t >> 5, lane = t & 31;
  int n0 = w * 32;
  v8f a00 = {}, a01 = {}, a10 = {}, a11 = {};
  for (int kk = 0; kk < 256; kk += 32) {
    v16bf al = load_fragA(At + kk, 256, lane);
    v16bf ah = load_fragA(At + 16 * 256 + kk, 256, lane);
    v16bf b0 = load_fragB(Wt + (size_t)n0 * 256 + kk, 256, lane);
    v16bf b1 = load_fragB(Wt + (size_t)(n0 + 16) * 256 + kk, 256, lane);
    a00 = wmma_bf16(al, b0, a00);
    a01 = wmma_bf16(al, b1, a01);
    a10 = wmma_bf16(ah, b0, a10);
    a11 = wmma_bf16(ah, b1, a11);
  }
  int nl = lane & 15, mb = (lane & 16) ? 8 : 0;
  v8f accs[4] = {a00, a01, a10, a11};
#pragma unroll
  for (int ti = 0; ti < 4; ++ti) {
    int msub = ti >> 1, nsub = ti & 1;
#pragma unroll
    for (int r = 0; r < 8; ++r) {
      int m_ = msub * 16 + mb + r;
      int col = n0 + nsub * 16 + nl;
      res[m_ * 256 + col] =
          accs[ti][r] + bfc[col] + query[(size_t)(R0 + m_) * 256 + col];
    }
  }
  __syncthreads();
  int row = t >> 3, seg = t & 7, cb = seg * 32;
  float s = 0.f, s2 = 0.f;
  for (int c = cb; c < cb + 32; ++c) {
    float v = res[row * 256 + c];
    s += v;
    s2 += v * v;
  }
  red[row * 8 + seg] = s;
  red2[row * 8 + seg] = s2;
  __syncthreads();
  if (t < 32) {
    float ts = 0.f, ts2 = 0.f;
    for (int i = 0; i < 8; ++i) {
      ts += red[t * 8 + i];
      ts2 += red2[t * 8 + i];
    }
    float m = ts / 256.f;
    float var = ts2 / 256.f - m * m;
    mu[t] = m;
    rs[t] = rsqrtf(var + 1e-5f);
  }
  __syncthreads();
  float m = mu[row], r_ = rs[row];
  for (int c = cb; c < cb + 32; ++c) {
    float v = (res[row * 256 + c] - m) * r_ * gamma[c] + beta[c];
    out[(size_t)(R0 + row) * 256 + c] = v;
  }
}

// ---------------- launcher ----------------
extern "C" void kernel_launch(void* const* d_in, const int* in_sizes, int n_in,
                              void* d_out, int out_size, void* d_ws, size_t ws_size,
                              hipStream_t stream) {
  const float* query = (const float*)d_in[0];
  const float* key_ = (const float*)d_in[1];
  const float* value = (const float*)d_in[2];
  const unsigned char* mask = (const unsigned char*)d_in[3];
  const float* relation = (const float*)d_in[4];
  const float* Wq = (const float*)d_in[5];
  const float* Wk = (const float*)d_in[6];
  const float* Wv = (const float*)d_in[7];
  const float* Wfc = (const float*)d_in[8];
  const float* bfc = (const float*)d_in[9];
  const float* gamma = (const float*)d_in[10];
  const float* beta = (const float*)d_in[11];

  char* ws = (char*)d_ws;
  __bf16* relT = (__bf16*)(ws + WS_RELT);
  __bf16* Wqt = (__bf16*)(ws + WS_WQT);
  __bf16* Wkt = (__bf16*)(ws + WS_WKT);
  __bf16* Wvt = (__bf16*)(ws + WS_WVT);
  __bf16* Wft = (__bf16*)(ws + WS_WFT);
  __bf16* Qws = (__bf16*)(ws + WS_Q);
  __bf16* Kws = (__bf16*)(ws + WS_K);
  __bf16* Vtws = (__bf16*)(ws + WS_VT);
  float* attws = (float*)(ws + WS_ATT);

  float* out = (float*)d_out;
  float* attn_out = out + (size_t)B_ * S_ * D_;

  // 0) transpose + bf16 conversions
  transpose_to_bf16<<<dim3(S_ / 32, S_ / 32, B_), dim3(32, 8), 0, stream>>>(
      relation, relT, S_, S_);
  transpose_to_bf16<<<dim3(8, 8, 1), dim3(32, 8), 0, stream>>>(Wq, Wqt, 256, 256);
  transpose_to_bf16<<<dim3(8, 8, 1), dim3(32, 8), 0, stream>>>(Wk, Wkt, 256, 256);
  transpose_to_bf16<<<dim3(8, 8, 1), dim3(32, 8), 0, stream>>>(Wv, Wvt, 256, 256);
  transpose_to_bf16<<<dim3(8, 8, 1), dim3(32, 8), 0, stream>>>(Wfc, Wft, 256, 256);

  // 1) QKV projections
  proj_kernel<<<dim3((B_ * S_) / 32, 2), 256, 0, stream>>>(query, Wqt, Qws, 0);
  proj_kernel<<<dim3((B_ * S_) / 32, 2), 256, 0, stream>>>(key_, Wkt, Kws, 0);
  proj_kernel<<<dim3((B_ * S_) / 32, 2), 256, 0, stream>>>(value, Wvt, Vtws, 1);

  // 2) fused attention core (260KB dynamic LDS, legal on CDNA5's 320KB WGP LDS)
  size_t smem = (size_t)(32 * 2048 * 2) * 2 + 32 * 32 * 2 +
                (size_t)(32 * 8 + 32 + 32) * 4;
  attn_kernel<<<dim3(S_ / 32, H_, B_), 256, smem, stream>>>(
      Qws, Kws, Vtws, relT, mask, attn_out, attws);

  // 3) FC + bias + residual + LayerNorm
  fc_ln_kernel<<<dim3((B_ * S_) / 32), 256, 0, stream>>>(attws, Wft, bfc, query,
                                                          gamma, beta, out);
}
